// CausalAttention_43087111913477
// MI455X (gfx1250) — matrix operand order repaired and checked by reading, not compiled, not run on hardware
//
#include <hip/hip_runtime.h>

// ---------------------------------------------------------------------------
// CDNA5 (gfx1250) causal attention block, bf16 WMMA pipeline, LDS-staged GEMMs
// with async global->LDS copies (ASYNCcnt) and software pipelining.
//
// Workspace layout (bytes), requires >= 48 MiB of d_ws:
//   [ 0MiB) xn      bf16 [4096,1024]       RMSNorm'd activations
//   [ 8MiB) wqkvT   bf16 [3072,1024]       w_qkv transposed (N,K)
//   [14MiB) woutT   bf16 [1024,1024]       w_out transposed (N,K)
//   [16MiB) q       bf16 [b,h,2048,64]     pre-scaled by 1/8
//   [24MiB) k       bf16 [b,h,2048,64]
//   [32MiB) vT      bf16 [b,h,64,2048]     v transposed per head
//   [40MiB) ao      bf16 [4096,1024]       attention output (pre out-proj)
// ---------------------------------------------------------------------------

typedef __bf16 bf16;
typedef __attribute__((ext_vector_type(16))) __bf16 v16bf;
typedef __attribute__((ext_vector_type(8)))  __bf16 v8bf;
typedef __attribute__((ext_vector_type(4)))  __bf16 v4bf;
typedef __attribute__((ext_vector_type(8)))  float  v8f;
typedef __attribute__((ext_vector_type(4)))  int    v4i;

#if __has_builtin(__builtin_amdgcn_global_load_async_to_lds_b128)
#define HAVE_ASYNC_LDS 1
// clang prints AS1 as "__device__" and AS3 as "__shared__" in HIP mode; the
// builtin takes (v4i AS1*, v4i AS3*, i32 offset, i32 cpol).
typedef __attribute__((address_space(1))) v4i gv4i;
typedef __attribute__((address_space(3))) v4i lv4i;
#else
#define HAVE_ASYNC_LDS 0
#endif

static __device__ __forceinline__ v8f vzero8() {
  v8f z;
#pragma unroll
  for (int i = 0; i < 8; ++i) z[i] = 0.0f;
  return z;
}

static __device__ __forceinline__ v8f wmma_bf16(v16bf a, v16bf b, v8f c) {
  // D = A(16x32 bf16) * B(32x16 bf16) + C(16x16 f32)
  return __builtin_amdgcn_wmma_f32_16x16x32_bf16(
      false, a, false, b, (short)0, c, false, false);
}

// A-operand loader (contiguous k order, same as ld_b32): rowp -> (row base + k base), row-major.
// Lanes 0-15 carry K {0..15}, lanes 16-31 carry K {16..31}.
static __device__ __forceinline__ v16bf ld_a32(const bf16* rowp, int lane) {
  const bf16* p = rowp + ((lane & 16) ? 16 : 0);
  v8bf lo = *(const v8bf*)(p);
  v8bf hi = *(const v8bf*)(p + 8);
  return __builtin_shufflevector(lo, hi, 0, 1, 2, 3, 4, 5, 6, 7,
                                 8, 9, 10, 11, 12, 13, 14, 15);
}

// B-operand loader: lane = column (L%16); lanes 0-15 K 0..15, 16-31 K 16..31.
static __device__ __forceinline__ v16bf ld_b32(const bf16* colp, int lane) {
  return *(const v16bf*)(colp + ((lane & 16) ? 16 : 0));
}

// Async copy of 32 contiguous bytes global -> LDS (ASYNCcnt tracked).
static __device__ __forceinline__ void cp_async32(bf16* l, const bf16* g) {
#if HAVE_ASYNC_LDS
  __builtin_amdgcn_global_load_async_to_lds_b128((gv4i*)g, (lv4i*)l, 0, 0);
  __builtin_amdgcn_global_load_async_to_lds_b128((gv4i*)(g + 8), (lv4i*)(l + 8),
                                                 0, 0);
#else
  *(v16bf*)l = *(const v16bf*)g;
#endif
}

static __device__ __forceinline__ void wait_async_zero() {
#if HAVE_ASYNC_LDS
  asm volatile("s_wait_asynccnt 0x0" ::: "memory");
#endif
}

// Stage a 64(M) x 32(K) bf16 A tile into LDS. 128 threads, 32B per thread.
static __device__ __forceinline__ void stage_a_tile(bf16* dst, const bf16* A,
                                                    int mbase, int k0, int tid) {
  const int row = tid >> 1;            // 0..63
  const int koff = (tid & 1) * 16;     // 0 or 16 elems
  cp_async32(dst + row * 32 + koff,
             A + (size_t)(mbase + row) * 1024 + k0 + koff);
}

// ---------------------------------------------------------------------------
// Shared GEMM core: block = 4 waves (128 thr). Block tile 64(M) x 128(N);
// wave computes 64x32 (acc[2][4]). A staged in LDS (double buffered, async),
// B register double-buffered from a pre-transposed [N,K] matrix. K = 1024.
// ---------------------------------------------------------------------------
static __device__ __forceinline__ void gemm_core(
    const bf16* __restrict__ A, const bf16* __restrict__ BT,
    bf16* lds /* [2][64*32] */, int mbase, int nbase, v8f acc[2][4]) {
  const int tid = threadIdx.x;
  const int lane = tid & 31;
#pragma unroll
  for (int nt = 0; nt < 2; ++nt)
#pragma unroll
    for (int mt = 0; mt < 4; ++mt) acc[nt][mt] = vzero8();

  const bf16* brow0 = BT + (size_t)(nbase + (lane & 15)) * 1024 +
                      ((lane & 16) ? 16 : 0);
  const bf16* brow1 = brow0 + (size_t)16 * 1024;

  stage_a_tile(lds, A, mbase, 0, tid);
  wait_async_zero();
  __syncthreads();

  v16bf b0 = *(const v16bf*)(brow0);
  v16bf b1 = *(const v16bf*)(brow1);

  int buf = 0;
  for (int k0 = 0; k0 < 1024; k0 += 32) {
    const int kn = (k0 + 32 < 1024) ? (k0 + 32) : k0;
    // Kick off next A tile (async) and next B registers while computing.
    if (k0 + 32 < 1024) stage_a_tile(lds + (buf ^ 1) * (64 * 32), A, mbase, kn, tid);
    __builtin_prefetch(brow0 + kn + 128, 0, 0);
    __builtin_prefetch(brow1 + kn + 128, 0, 0);
    v16bf nb0 = *(const v16bf*)(brow0 + kn);
    v16bf nb1 = *(const v16bf*)(brow1 + kn);

    const bf16* la = lds + buf * (64 * 32);
#pragma unroll
    for (int mt = 0; mt < 4; ++mt) {
      v16bf a = ld_a32(la + (((lane & 15) + 16 * mt) * 32), lane);
      acc[0][mt] = wmma_bf16(a, b0, acc[0][mt]);
      acc[1][mt] = wmma_bf16(a, b1, acc[1][mt]);
    }
    b0 = nb0;
    b1 = nb1;
    wait_async_zero();
    __syncthreads();
    buf ^= 1;
  }
}

// ---------------------------------------------------------------------------
// Prep: fp32 [K,N] -> bf16 [N,K] transpose (weights).
// ---------------------------------------------------------------------------
__global__ void transpose_to_bf16_kernel(const float* __restrict__ src,
                                         bf16* __restrict__ dst,
                                         int K, int N) {
  int idx = blockIdx.x * 256 + threadIdx.x;
  if (idx >= K * N) return;
  int n = idx / K;
  int k = idx - n * K;
  dst[idx] = (bf16)src[(size_t)k * N + n];
}

// ---------------------------------------------------------------------------
// RMSNorm: per row of 1024; xn = x / max(||x||,1e-12) * 32 * gamma (bf16 out)
// ---------------------------------------------------------------------------
__global__ void rmsnorm_kernel(const float* __restrict__ x,
                               const float* __restrict__ gamma,
                               bf16* __restrict__ xn) {
  const int row = blockIdx.x;
  const int t = threadIdx.x;  // 256 threads, 4 elems each
  const float4 v = ((const float4*)(x + (size_t)row * 1024))[t];
  float ss = v.x * v.x + v.y * v.y + v.z * v.z + v.w * v.w;
#pragma unroll
  for (int off = 16; off > 0; off >>= 1) ss += __shfl_xor(ss, off, 32);
  __shared__ float part[8];
  if ((t & 31) == 0) part[t >> 5] = ss;
  __syncthreads();
  float tot = 0.0f;
#pragma unroll
  for (int i = 0; i < 8; ++i) tot += part[i];
  const float s = 32.0f / fmaxf(sqrtf(tot), 1e-12f);
  const float4 g = ((const float4*)gamma)[t];
  v4bf o;
  o[0] = (bf16)(v.x * g.x * s);
  o[1] = (bf16)(v.y * g.y * s);
  o[2] = (bf16)(v.z * g.z * s);
  o[3] = (bf16)(v.w * g.w * s);
  ((v4bf*)xn)[(size_t)row * 256 + t] = o;
}

// ---------------------------------------------------------------------------
// QKV GEMM: [4096,3072] = xn[4096,1024] @ w_qkv (B pre-transposed).
// Grid (24, 64), block 128. Epilogue scatters q (scaled) / k / vT.
// ---------------------------------------------------------------------------
__global__ void gemm_qkv_kernel(const bf16* __restrict__ A,
                                const bf16* __restrict__ BT,
                                bf16* __restrict__ q,
                                bf16* __restrict__ kk,
                                bf16* __restrict__ vT) {
  __shared__ bf16 lds_a[2][64 * 32];
  const int lane = threadIdx.x & 31;
  const int wave = threadIdx.x >> 5;
  const int mbase = blockIdx.y * 64;
  const int nbase = blockIdx.x * 128 + wave * 32;
  v8f acc[2][4];
  gemm_core(A, BT, &lds_a[0][0], mbase, nbase, acc);

#pragma unroll
  for (int nt = 0; nt < 2; ++nt) {
    const int nsub = nbase + 16 * nt;
    const int seg = nsub >> 10;  // 0=q 1=k 2=v
    const int h = (nsub & 1023) >> 6;
    const int dcol = (nsub & 63) + (lane & 15);
#pragma unroll
    for (int mt = 0; mt < 4; ++mt) {
#pragma unroll
      for (int r = 0; r < 8; ++r) {
        const int m = mbase + 16 * mt + r + ((lane & 16) ? 8 : 0);
        const int b = m >> 11;
        const int pos = m & 2047;
        const size_t bh = (size_t)(b * 16 + h);
        const float val = acc[nt][mt][r];
        if (seg == 0) {
          q[(bh * 2048 + pos) * 64 + dcol] = (bf16)(val * 0.125f);
        } else if (seg == 1) {
          kk[(bh * 2048 + pos) * 64 + dcol] = (bf16)val;
        } else {
          vT[(bh * 64 + dcol) * 2048 + pos] = (bf16)val;
        }
      }
    }
  }
}

// ---------------------------------------------------------------------------
// Flash causal attention. One wave per 16-query tile; online softmax over
// j-blocks of 32. sim^T = K @ Q^T so the softmax reduce is in-lane + xor16.
// All K/V tiles for a block are loaded up-front so the V loads overlap the
// softmax VALU work.
// ---------------------------------------------------------------------------
__global__ void attn_kernel(const bf16* __restrict__ Q,
                            const bf16* __restrict__ K,
                            const bf16* __restrict__ VT,
                            bf16* __restrict__ O) {
  const int lane = threadIdx.x & 31;
  const int wave = threadIdx.x >> 5;
  const int qbase = (blockIdx.x * 4 + wave) * 16;
  const int h = blockIdx.y, b = blockIdx.z;
  const size_t bh = (size_t)(b * 16 + h);
  const bf16* q = Q + bh * (2048 * 64);
  const bf16* k = K + bh * (2048 * 64);
  const bf16* vt = VT + bh * (64 * 2048);

  // Q^T as B operand, resident for the whole loop (two K=32 chunks of d=64).
  const bf16* qrow = q + (size_t)(qbase + (lane & 15)) * 64;
  const v16bf qb0 = ld_b32(qrow, lane);
  const v16bf qb1 = ld_b32(qrow + 32, lane);

  v8f o[4];
#pragma unroll
  for (int t = 0; t < 4; ++t) o[t] = vzero8();
  float m = -1e30f, l = 0.0f;

  const int iq = qbase + (lane & 15);
  const int nblocks = (qbase + 16 + 31) >> 5;
  const bool hiHalf = (lane & 16) != 0;

  for (int jblk = 0; jblk < nblocks; ++jblk) {
    const int jb = jblk * 32;
    // Batch-issue all loads for this block: 4 K-subtiles + 4 V-subtiles.
    const bf16* kr0 = k + (size_t)(jb + (lane & 15)) * 64;
    v16bf a00 = ld_a32(kr0, lane);
    v16bf a01 = ld_a32(kr0 + 32, lane);
    v16bf a10 = ld_a32(kr0 + 16 * 64, lane);
    v16bf a11 = ld_a32(kr0 + 16 * 64 + 32, lane);
    v16bf va[4];
#pragma unroll
    for (int t = 0; t < 4; ++t)
      va[t] = ld_a32(vt + (size_t)(16 * t + (lane & 15)) * 2048 + jb, lane);

    v8f s0 = vzero8(), s1 = vzero8();
    s0 = wmma_bf16(a00, qb0, s0);
    s0 = wmma_bf16(a01, qb1, s0);
    s1 = wmma_bf16(a10, qb0, s1);
    s1 = wmma_bf16(a11, qb1, s1);

    // Causal mask (only blocks touching the diagonal).
    if (jb + 31 > qbase) {
      const int j0 = jb + (hiHalf ? 8 : 0);
#pragma unroll
      for (int r = 0; r < 8; ++r) {
        if (j0 + r > iq) s0[r] = -1e30f;
        if (j0 + 16 + r > iq) s1[r] = -1e30f;
      }
    }
    // Block row-max for this query column.
    float bm = -1e30f;
#pragma unroll
    for (int r = 0; r < 8; ++r) bm = fmaxf(bm, fmaxf(s0[r], s1[r]));
    bm = fmaxf(bm, __shfl_xor(bm, 16, 32));
    const float mnew = fmaxf(m, bm);
    const float alpha = __expf(m - mnew);
    m = mnew;

    float p0[8], p1[8];
    float lsum = 0.0f;
#pragma unroll
    for (int r = 0; r < 8; ++r) {
      p0[r] = __expf(s0[r] - mnew);
      p1[r] = __expf(s1[r] - mnew);
      lsum += p0[r] + p1[r];
    }
    lsum += __shfl_xor(lsum, 16, 32);
    l = l * alpha + lsum;

    // Repack P^T into B-operand layout (lanes 0-15: j=jb..+15; 16-31: +16).
    v16bf bp;
#pragma unroll
    for (int r = 0; r < 8; ++r) {
      const float o0 = __shfl_xor(p0[r], 16, 32);
      const float o1 = __shfl_xor(p1[r], 16, 32);
      bp[r]     = (bf16)(hiHalf ? o1 : p0[r]);
      bp[8 + r] = (bf16)(hiHalf ? p1[r] : o0);
    }

    // out^T[64,16] += V^T[c, j] @ P^T[j, i], with online rescale by alpha.
#pragma unroll
    for (int t = 0; t < 4; ++t) {
#pragma unroll
      for (int r = 0; r < 8; ++r) o[t][r] = o[t][r] * alpha;
      o[t] = wmma_bf16(va[t], bp, o[t]);
    }
  }

  const float inv = 1.0f / l;
  // out^T element (c = 16t + r + 8*hi, i = lane%16) -> ao[b*2048+i, h*64+c]
  const size_t rowbase =
      ((size_t)b * 2048 + qbase + (lane & 15)) * 1024 + (size_t)h * 64;
  const int choff = hiHalf ? 8 : 0;
#pragma unroll
  for (int t = 0; t < 4; ++t) {
    v8bf st;
#pragma unroll
    for (int r = 0; r < 8; ++r) st[r] = (bf16)(o[t][r] * inv);
    *(v8bf*)(O + rowbase + 16 * t + choff) = st;
  }
}

// ---------------------------------------------------------------------------
// Output GEMM: out[4096,1024] = ao[4096,1024] @ w_out (B pre-transposed).
// Grid (8, 64), block 128.
// ---------------------------------------------------------------------------
__global__ void gemm_out_kernel(const bf16* __restrict__ A,
                                const bf16* __restrict__ BT,
                                float* __restrict__ out) {
  __shared__ bf16 lds_a[2][64 * 32];
  const int lane = threadIdx.x & 31;
  const int wave = threadIdx.x >> 5;
  const int mbase = blockIdx.y * 64;
  const int nbase = blockIdx.x * 128 + wave * 32;
  v8f acc[2][4];
  gemm_core(A, BT, &lds_a[0][0], mbase, nbase, acc);

#pragma unroll
  for (int nt = 0; nt < 2; ++nt) {
    const int col = nbase + 16 * nt + (lane & 15);
#pragma unroll
    for (int mt = 0; mt < 4; ++mt) {
#pragma unroll
      for (int r = 0; r < 8; ++r) {
        const int mrow = mbase + 16 * mt + r + ((lane & 16) ? 8 : 0);
        out[(size_t)mrow * 1024 + col] = acc[nt][mt][r];
      }
    }
  }
}

// ---------------------------------------------------------------------------
extern "C" void kernel_launch(void* const* d_in, const int* in_sizes, int n_in,
                              void* d_out, int out_size, void* d_ws,
                              size_t ws_size, hipStream_t stream) {
  const float* x = (const float*)d_in[0];      // [2,2048,1024]
  const float* gamma = (const float*)d_in[1];  // [1024]
  const float* w_qkv = (const float*)d_in[2];  // [1024,3072]
  const float* w_out = (const float*)d_in[3];  // [1024,1024]
  float* out = (float*)d_out;                  // [2,2048,1024]

  char* ws = (char*)d_ws;
  bf16* xn    = (bf16*)(ws + (size_t)0);
  bf16* wqkvT = (bf16*)(ws + ((size_t)8 << 20));
  bf16* woutT = (bf16*)(ws + ((size_t)14 << 20));
  bf16* q     = (bf16*)(ws + ((size_t)16 << 20));
  bf16* k     = (bf16*)(ws + ((size_t)24 << 20));
  bf16* vT    = (bf16*)(ws + ((size_t)32 << 20));
  bf16* ao    = (bf16*)(ws + ((size_t)40 << 20));

  transpose_to_bf16_kernel<<<(1024 * 3072 + 255) / 256, 256, 0, stream>>>(
      w_qkv, wqkvT, 1024, 3072);
  transpose_to_bf16_kernel<<<(1024 * 1024 + 255) / 256, 256, 0, stream>>>(
      w_out, woutT, 1024, 1024);
  rmsnorm_kernel<<<4096, 256, 0, stream>>>(x, gamma, xn);
  gemm_qkv_kernel<<<dim3(24, 64), 128, 0, stream>>>(xn, wqkvT, q, k, vT);
  attn_kernel<<<dim3(32, 16, 2), 128, 0, stream>>>(q, k, vT, ao);
  gemm_out_kernel<<<dim3(8, 64), 128, 0, stream>>>(ao, woutT, out);
}
